// LFSRCompressedLayer_62302795595847
// MI455X (gfx1250) — matrix operand order repaired and checked by reading, not compiled, not run on hardware
//
#include <hip/hip_runtime.h>
#include <stdint.h>

#define B_DIM   512
#define IN_DIM  4096
#define OUT_DIM 4096

typedef __attribute__((ext_vector_type(16))) __bf16 v16bf;
typedef __attribute__((ext_vector_type(8)))  float  v8f;

__device__ __forceinline__ uint16_t f2bf(float f) {
    uint32_t u = __float_as_uint(f);
    // round-to-nearest-even truncation to bf16
    return (uint16_t)((u + 0x7FFFu + ((u >> 16) & 1u)) >> 16);
}

// Phase 1: W[o*IN+i] = bf16( dot(U[idx][0..3], t_hat) ), t_hat = q/7 * 2^exp
// Streams 256 MB of U (the roofline floor), writes 32 MB bf16 W.
__global__ void __launch_bounds__(256) reconstruct_w_kernel(
        const float4* __restrict__ U, const float* __restrict__ q,
        const int* __restrict__ expn, uint16_t* __restrict__ wbf) {
    const float s  = exp2f((float)expn[0]) * (1.0f / 7.0f);
    const float t0 = q[0] * s, t1 = q[1] * s, t2 = q[2] * s, t3 = q[3] * s;

    size_t base = ((size_t)blockIdx.x * blockDim.x + threadIdx.x) * 8;
    uint16_t tmp[8];
#pragma unroll
    for (int e = 0; e < 8; ++e) {
        float4 u4 = U[base + e];
        tmp[e] = f2bf(u4.x * t0 + u4.y * t1 + u4.z * t2 + u4.w * t3);
    }
    uint4 pk;
    pk.x = (uint32_t)tmp[0] | ((uint32_t)tmp[1] << 16);
    pk.y = (uint32_t)tmp[2] | ((uint32_t)tmp[3] << 16);
    pk.z = (uint32_t)tmp[4] | ((uint32_t)tmp[5] << 16);
    pk.w = (uint32_t)tmp[6] | ((uint32_t)tmp[7] << 16);
    *reinterpret_cast<uint4*>(wbf + base) = pk;
}

// Phase 2: x (f32) -> bf16
__global__ void __launch_bounds__(256) convert_x_kernel(
        const float4* __restrict__ x, uint16_t* __restrict__ xbf) {
    size_t base = ((size_t)blockIdx.x * blockDim.x + threadIdx.x) * 8;
    float4 a = x[base / 4];
    float4 b = x[base / 4 + 1];
    uint4 pk;
    pk.x = (uint32_t)f2bf(a.x) | ((uint32_t)f2bf(a.y) << 16);
    pk.y = (uint32_t)f2bf(a.z) | ((uint32_t)f2bf(a.w) << 16);
    pk.z = (uint32_t)f2bf(b.x) | ((uint32_t)f2bf(b.y) << 16);
    pk.w = (uint32_t)f2bf(b.z) | ((uint32_t)f2bf(b.w) << 16);
    *reinterpret_cast<uint4*>(xbf + base) = pk;
}

// Phase 3: y[m][n] = sum_k xbf[m][k] * wbf[n][k] + bias[n]
// Wave tile 64(M) x 64(N): acc[4][4] 16x16 WMMA f32 accumulators (128 VGPRs).
// Per k-tile: 16 b128 loads -> 16 WMMAs (1:1 issue ratio, 4x frag reuse).
// Block = 8 waves arranged 2(M) x 4(N) -> 128 x 256 block tile; 64 blocks total,
// so total L2 read traffic ~= 64 * (1MB x + 2MB W) = 192 MB (W+x fully L2-resident).
// A fragment (16x32 bf16): lane l<16 holds M=l, K = hi*8 + {0..7, 16..23}
// B fragment (32x16 bf16): lane l<16 holds N=l, K = hi*8 + {0..7, 16..23}
//   -> W rows are K-contiguous, so B loads are two 16B loads.
__global__ void __launch_bounds__(256) gemm_bf16_kernel(
        const uint16_t* __restrict__ xbf, const uint16_t* __restrict__ wbf,
        const float* __restrict__ bias, float* __restrict__ out) {
    const int lane = threadIdx.x & 31;
    const int wave = threadIdx.x >> 5;
    const int l16  = lane & 15;
    const int hi   = lane >> 4;

    const int waveM = wave & 1;
    const int waveN = wave >> 1;
    const int Mbase = blockIdx.y * 128 + waveM * 64;
    const int Nbase = blockIdx.x * 256 + waveN * 64;

    v8f acc[4][4];
    const v8f vzero = {0.f, 0.f, 0.f, 0.f, 0.f, 0.f, 0.f, 0.f};
#pragma unroll
    for (int mt = 0; mt < 4; ++mt)
#pragma unroll
        for (int nt = 0; nt < 4; ++nt)
            acc[mt][nt] = vzero;

    union Frag { v16bf v; uint4 u[2]; };

    for (int k0 = 0; k0 < IN_DIM; k0 += 32) {
        Frag a[4];
#pragma unroll
        for (int mt = 0; mt < 4; ++mt) {
            const uint16_t* p = xbf + (size_t)(Mbase + mt * 16 + l16) * IN_DIM
                                    + k0 + hi * 8;
            a[mt].u[0] = *reinterpret_cast<const uint4*>(p);       // K = base..base+7
            a[mt].u[1] = *reinterpret_cast<const uint4*>(p + 16);  // K = base+16..base+23
        }
        Frag bf[4];
#pragma unroll
        for (int nt = 0; nt < 4; ++nt) {
            const uint16_t* p = wbf + (size_t)(Nbase + nt * 16 + l16) * IN_DIM
                                    + k0 + hi * 8;
            const uint4* pv = reinterpret_cast<const uint4*>(p);
            bf[nt].u[0] = pv[0];   // K = hi*8 + 0..7
            bf[nt].u[1] = pv[2];   // K = hi*8 + 16..23
        }
#pragma unroll
        for (int mt = 0; mt < 4; ++mt)
#pragma unroll
            for (int nt = 0; nt < 4; ++nt)
                acc[mt][nt] = __builtin_amdgcn_wmma_f32_16x16x32_bf16(
                    false, a[mt].v, false, bf[nt].v,
                    (short)0, acc[mt][nt], false, false);
    }

    // C/D layout: element r -> lanes 0-15: (M = r, N = lane); lanes 16-31: (M = r+8, N = lane-16)
#pragma unroll
    for (int mt = 0; mt < 4; ++mt) {
#pragma unroll
        for (int nt = 0; nt < 4; ++nt) {
            const int col  = Nbase + nt * 16 + l16;
            const float bv = bias[col];
            const int row0 = Mbase + mt * 16 + hi * 8;
#pragma unroll
            for (int r = 0; r < 8; ++r)
                out[(size_t)(row0 + r) * OUT_DIM + col] = acc[mt][nt][r] + bv;
        }
    }
}

extern "C" void kernel_launch(void* const* d_in, const int* in_sizes, int n_in,
                              void* d_out, int out_size, void* d_ws, size_t ws_size,
                              hipStream_t stream) {
    const float4* x    = (const float4*)d_in[0];
    const float4* U    = (const float4*)d_in[1];
    const float*  q    = (const float*) d_in[2];
    const float*  b    = (const float*) d_in[3];
    const int*    expn = (const int*)   d_in[4];
    float* out = (float*)d_out;

    uint16_t* wbf = (uint16_t*)d_ws;                                   // 32 MB
    uint16_t* xbf = (uint16_t*)((char*)d_ws + (size_t)OUT_DIM * IN_DIM * 2); // 4 MB

    // W reconstruction: 16.7M elems, 8 per thread
    reconstruct_w_kernel<<<(OUT_DIM * (size_t)IN_DIM) / 8 / 256, 256, 0, stream>>>(
        U, q, expn, wbf);
    // x conversion: 2M elems, 8 per thread
    convert_x_kernel<<<(B_DIM * (size_t)IN_DIM) / 8 / 256, 256, 0, stream>>>(x, xbf);
    // GEMM: grid = (N tiles, M tiles)
    dim3 grid(OUT_DIM / 256, B_DIM / 128);
    gemm_bf16_kernel<<<grid, 256, 0, stream>>>(xbf, wbf, b, out);
}
